// MoE_12970801234427
// MI455X (gfx1250) — compile-verified, matrix operand order checked by reading, not run on hardware
//
#include <hip/hip_runtime.h>
#include <hip/hip_bf16.h>

// ---------------------------------------------------------------------------
// MoE (top-2 of 8 experts), N=8192 tokens, C=1024, HIDDEN=2752 (2H=5504).
// Only selected experts contribute (others have weight exactly 0): gather
// tokens per expert, run bf16 WMMA GEMMs on the 2N=16384 assignments only
// (~277 GFLOP vs ~1.1 TFLOP dense).  x-tile gather uses the CDNA5 Tensor
// Data Mover (gather mode, 16 row indices) when the builtin is available.
// ---------------------------------------------------------------------------

#define TOK_N  8192
#define EMB    1024
#define NEXP   8
#define HID    2752
#define HID2   5504
#define CHUNK  128

typedef __attribute__((ext_vector_type(16))) __bf16 v16bf;
typedef __attribute__((ext_vector_type(8)))  __bf16 v8bf;
typedef __attribute__((ext_vector_type(4)))  __bf16 v4bf;
typedef __attribute__((ext_vector_type(8)))  float  v8f;
typedef __attribute__((ext_vector_type(4)))  unsigned int u32x4;
typedef __attribute__((ext_vector_type(8)))  int          i32x8;
typedef __attribute__((ext_vector_type(4)))  int          i32x4;

#if defined(__has_builtin)
#if __has_builtin(__builtin_amdgcn_tensor_load_to_lds) && \
    __has_builtin(__builtin_amdgcn_s_wait_tensorcnt)
#define USE_TDM 1
#endif
#endif
#ifndef USE_TDM
#define USE_TDM 0
#endif

#if defined(__has_builtin)
#if __has_builtin(__builtin_amdgcn_rcpf)
#define FAST_RCP(x) __builtin_amdgcn_rcpf(x)
#endif
#endif
#ifndef FAST_RCP
#define FAST_RCP(x) (1.0f / (x))
#endif

// Load a 16x32 bf16 fragment (A/B operand layout per CDNA5 ISA 7.12.2:
// lanes 0-15 carry row=lane with K = {0..7, 16..23}; lanes 16-31 carry
// row=lane-16 with K = {8..15, 24..31}).  Two 16B ds loads per lane.
static __device__ inline v16bf ld_frag(const __bf16* base, int stride, int lane, int kb) {
    const int half = (lane >> 4) & 1;
    const int r    = lane & 15;
    const __bf16* p = base + r * stride + kb + half * 8;
    v8bf lo = *(const v8bf*)(p);
    v8bf hi = *(const v8bf*)(p + 16);
    v16bf o;
#pragma unroll
    for (int i = 0; i < 8; ++i) { o[i] = lo[i]; o[i + 8] = hi[i]; }
    return o;
}

static __device__ inline v8f wmma_bf16(v16bf a, v16bf b, v8f c) {
    // D(f32 16x16) = A(bf16 16x32) * B(bf16 32x16) + C
    return __builtin_amdgcn_wmma_f32_16x16x32_bf16(false, a, false, b,
                                                   (short)0, c, false, false);
}

// ---------------------------------------------------------------------------
// K1: router.  One wave per token; wave32 shuffle reduction.
// ---------------------------------------------------------------------------
__global__ __launch_bounds__(256) void k1_router(
        const float* __restrict__ x, const float* __restrict__ Wr,
        float* __restrict__ probs, float* __restrict__ zterm,
        int* __restrict__ topi, float* __restrict__ topw) {
    __shared__ float sWr[EMB * NEXP];   // 32 KB
    const int tid = threadIdx.x;
    for (int i = tid; i < EMB * NEXP; i += 256) sWr[i] = Wr[i];
    __syncthreads();

    const int lane = tid & 31;
    const int t    = blockIdx.x * 8 + (tid >> 5);
    const float* xr = x + (size_t)t * EMB;

    float acc[NEXP];
#pragma unroll
    for (int e = 0; e < NEXP; ++e) acc[e] = 0.f;

    for (int i = lane; i < EMB; i += 32) {
        const float xv = xr[i];
        const int   b  = i * NEXP;
#pragma unroll
        for (int e = 0; e < NEXP; ++e) acc[e] = fmaf(xv, sWr[b + e], acc[e]);
    }
#pragma unroll
    for (int e = 0; e < NEXP; ++e)
        for (int o = 16; o > 0; o >>= 1) acc[e] += __shfl_xor(acc[e], o, 32);

    if (lane == 0) {
        float m = acc[0];
#pragma unroll
        for (int e = 1; e < NEXP; ++e) m = fmaxf(m, acc[e]);
        float p[NEXP], s = 0.f;
#pragma unroll
        for (int e = 0; e < NEXP; ++e) { p[e] = __expf(acc[e] - m); s += p[e]; }
        const float lse = m + __logf(s);
        zterm[t] = lse * lse;
        const float inv = FAST_RCP(s);
#pragma unroll
        for (int e = 0; e < NEXP; ++e) { p[e] *= inv; probs[t * NEXP + e] = p[e]; }
        int i0 = 0;
#pragma unroll
        for (int e = 1; e < NEXP; ++e) if (p[e] > p[i0]) i0 = e;
        int i1 = (i0 == 0) ? 1 : 0;
#pragma unroll
        for (int e = 0; e < NEXP; ++e) if (e != i0 && p[e] > p[i1]) i1 = e;
        const float sw = FAST_RCP(p[i0] + p[i1] + 1e-9f);
        topi[2 * t]     = i0;  topi[2 * t + 1] = i1;
        topw[2 * t]     = p[i0] * sw;
        topw[2 * t + 1] = p[i1] * sw;
    }
}

// ---------------------------------------------------------------------------
// K2: deterministic tree reductions for l_aux / z_loss; 16-row-padded
// per-expert offsets; zeroes fill cursors.
// ---------------------------------------------------------------------------
static __device__ inline float block_red_256(float v, float* sred) {
    const int tid = threadIdx.x;
    sred[tid] = v; __syncthreads();
#pragma unroll
    for (int s = 128; s > 0; s >>= 1) {
        if (tid < s) sred[tid] += sred[tid + s];
        __syncthreads();
    }
    const float r = sred[0];
    __syncthreads();
    return r;
}

__global__ __launch_bounds__(256) void k2_reduce(
        const float* __restrict__ probs, const float* __restrict__ zterm,
        const int* __restrict__ topi, float* __restrict__ out_tail,
        int* __restrict__ off, int* __restrict__ fill) {
    __shared__ float sred[256];
    const int tid = threadIdx.x;

    float imp[NEXP]; int c1[NEXP]; int ac[NEXP];
#pragma unroll
    for (int e = 0; e < NEXP; ++e) { imp[e] = 0.f; c1[e] = 0; ac[e] = 0; }
    float zs = 0.f;
    for (int t = tid; t < TOK_N; t += 256) {
#pragma unroll
        for (int e = 0; e < NEXP; ++e) imp[e] += probs[t * NEXP + e];
        zs += zterm[t];
        const int i0 = topi[2 * t], i1 = topi[2 * t + 1];
        c1[i0]++; ac[i0]++; ac[i1]++;
    }

    float impS[NEXP], ldS[NEXP], acS[NEXP];
    for (int e = 0; e < NEXP; ++e) {
        impS[e] = block_red_256(imp[e], sred);
        ldS[e]  = block_red_256((float)c1[e], sred);
        acS[e]  = block_red_256((float)ac[e], sred);
    }
    const float zsum = block_red_256(zs, sred);

    if (tid == 0) {
        float la = 0.f;
        for (int e = 0; e < NEXP; ++e)
            la += (ldS[e] / (float)TOK_N) * (impS[e] / (float)TOK_N);
        out_tail[0] = (float)NEXP * la;        // l_aux
        out_tail[1] = zsum / (float)TOK_N;     // z_loss
        int o = 0;
        for (int e = 0; e < NEXP; ++e) {
            off[e] = o;
            const int a = (int)acS[e];
            o += ((a + 15) >> 4) << 4;         // pad each expert to 16 rows
            fill[e] = 0;
        }
        off[NEXP] = o;
    }
}

// ---------------------------------------------------------------------------
// K3: scatter token ids + gate weights into per-expert lists (int atomics;
// y stays bitwise deterministic: rows are independent, 2 commutative adds).
// ---------------------------------------------------------------------------
__global__ __launch_bounds__(256) void k3_scatter(
        const int* __restrict__ topi, const float* __restrict__ topw,
        const int* __restrict__ off, int* __restrict__ fill,
        int* __restrict__ atok, float* __restrict__ aw) {
    const int t = blockIdx.x * 256 + threadIdx.x;
    if (t >= TOK_N) return;
#pragma unroll
    for (int j = 0; j < 2; ++j) {
        const int e = topi[2 * t + j];
        const int p = atomicAdd(&fill[e], 1);
        const int s = off[e] + p;
        atok[s] = t;
        aw[s]   = topw[2 * t + j];
    }
}

// ---------------------------------------------------------------------------
// K4: fused expert FFN.  One block = 16 assignments of one expert,
// 256 threads = 8 waves.  TDM gather-mode DMA pulls the 16 gathered x rows
// (fp32) into LDS in one tensor_load_to_lds; one pass converts to a bf16
// A-tile; the fp32 region is then recycled for the ACT tile and per-wave
// B staging.  fc output is produced in 128-col chunks, SwiGLU'd (rcp, not
// div) into LDS, and consumed by the proj GEMM whose per-wave 16x128 fp32
// accumulators persist across chunks.  All GEMM math is
// v_wmma_f32_16x16x32_bf16; epilogue = weight-scaled global f32 atomics.
// ---------------------------------------------------------------------------
#define A_STR   1032
#define ACT_STR 136
#define B_STR   40
// dynamic LDS layout (bytes):
//   [0,      65536)  fp32 x tile (TDM dest); recycled: ACT @0 (4352B),
//                    Blds @8192 (20480B)
//   [65536,  98560)  bf16 A tile 16 x A_STR
//   [98560,  98688)  s_tok[16], s_w[16]
#define SM_ALDS 65536
#define SM_BL   8192
#define SM_TOK  98560
#define SM_TOTAL 98688

__global__ __launch_bounds__(256) void k4_ffn(
        const float* __restrict__ x, const float* __restrict__ Wfc,
        const float* __restrict__ Wproj,
        const int* __restrict__ atok, const float* __restrict__ aw,
        const int* __restrict__ off, float* __restrict__ y) {
    extern __shared__ __align__(16) char smem[];
    float* Xf32 = (float*)smem;                                        // [16*1024]
    __bf16 (*ACT)[ACT_STR]       = (__bf16 (*)[ACT_STR])smem;          // recycled
    __bf16 (*Blds)[2][16][B_STR] = (__bf16 (*)[2][16][B_STR])(smem + SM_BL);
    __bf16 (*Alds)[A_STR]        = (__bf16 (*)[A_STR])(smem + SM_ALDS);
    int*   s_tok = (int*)(smem + SM_TOK);
    float* s_w   = (float*)(smem + SM_TOK + 64);

    const int row0 = blockIdx.x * 16;
    if (row0 >= off[NEXP]) return;                 // block-uniform exit
    int e = 0;
#pragma unroll
    for (int i = 1; i < NEXP; ++i) if (row0 >= off[i]) e = i;

    const int tid  = threadIdx.x;
    const int lane = tid & 31;
    const int w    = tid >> 5;
    const int half = lane >> 4;
    const int nl   = lane & 15;

    if (tid < 16) { s_tok[tid] = atok[row0 + tid]; s_w[tid] = aw[row0 + tid]; }
    __syncthreads();

#if USE_TDM
    // ---- TDM gather: 16 token rows x 1024 fp32 -> LDS in one DMA ----
    if (w == 0) {                                  // one wave issues the DMA
        const unsigned long long ga = (unsigned long long)(uintptr_t)x;
        u32x4 g0 = { 0x80000001u,                  // count=1, gather_mode, idx16
                     (unsigned)(uintptr_t)Xf32,    // lds byte offset
                     (unsigned)(ga & 0xFFFFFFFFu),
                     (unsigned)((ga >> 32) & 0x01FFFFFFu) | 0x80000000u }; // type=2
        i32x8 g1 = { 0x00020000,                   // data_size = 4B
                     (int)((EMB & 0xFFFF) << 16),  // tensor_dim0 lo
                     (int)((EMB >> 16) | ((TOK_N & 0xFFFF) << 16)),
                     (int)((TOK_N >> 16) | (EMB << 16)),   // tile_dim0 = 1024
                     16,                           // tile_dim1 = 16 gather rows
                     EMB,                          // tensor_dim0_stride
                     0, 0 };
        i32x4 g2 = { (s_tok[0]  & 0xFFFF) | (s_tok[1]  << 16),
                     (s_tok[2]  & 0xFFFF) | (s_tok[3]  << 16),
                     (s_tok[4]  & 0xFFFF) | (s_tok[5]  << 16),
                     (s_tok[6]  & 0xFFFF) | (s_tok[7]  << 16) };
        i32x4 g3 = { (s_tok[8]  & 0xFFFF) | (s_tok[9]  << 16),
                     (s_tok[10] & 0xFFFF) | (s_tok[11] << 16),
                     (s_tok[12] & 0xFFFF) | (s_tok[13] << 16),
                     (s_tok[14] & 0xFFFF) | (s_tok[15] << 16) };
#if __clang_major__ >= 23
        i32x8 gz = { 0, 0, 0, 0, 0, 0, 0, 0 };
        __builtin_amdgcn_tensor_load_to_lds(g0, g1, g2, g3, gz, 0);
#else
        __builtin_amdgcn_tensor_load_to_lds(g0, g1, g2, g3, 0);
#endif
        __builtin_amdgcn_s_wait_tensorcnt(0);
    }
    __syncthreads();
    // one-pass fp32 -> bf16 conversion into the padded A tile
    for (int j = tid; j < 16 * EMB / 4; j += 256) {
        const int lin = j * 4;
        const int r = lin >> 10, c = lin & (EMB - 1);
        const float4 v = *(const float4*)&Xf32[lin];
        v4bf b; b[0] = (__bf16)v.x; b[1] = (__bf16)v.y;
                b[2] = (__bf16)v.z; b[3] = (__bf16)v.w;
        *(v4bf*)&Alds[r][c] = b;
    }
#else
    // fallback: direct gather from global, float4 loads, packed b64 stores
    for (int j = tid; j < 16 * EMB / 4; j += 256) {
        const int lin = j * 4;
        const int r = lin >> 10, c = lin & (EMB - 1);
        const float4 v = *(const float4*)&x[(size_t)s_tok[r] * EMB + c];
        v4bf b; b[0] = (__bf16)v.x; b[1] = (__bf16)v.y;
                b[2] = (__bf16)v.z; b[3] = (__bf16)v.w;
        *(v4bf*)&Alds[r][c] = b;
    }
#endif
    __syncthreads();

    const float* Wfce = Wfc   + (size_t)e * EMB * HID2;
    const float* Wpre = Wproj + (size_t)e * HID * EMB;

    v8f accD[8];                                   // 128 output cols per wave
#pragma unroll
    for (int nt = 0; nt < 8; ++nt)
#pragma unroll
        for (int i = 0; i < 8; ++i) accD[nt][i] = 0.f;

    for (int kc = 0; kc < HID; kc += CHUNK) {
        const int ccols = (HID - kc < CHUNK) ? (HID - kc) : CHUNK;

        // ---- fc + SwiGLU: wave w produces act columns [kc+w*16, +16) ----
        if (w * 16 < ccols) {                      // wave-uniform branch
            const int cg = kc + w * 16;            // gate col; val col = cg+HID
            v8f ag, av;
#pragma unroll
            for (int i = 0; i < 8; ++i) { ag[i] = 0.f; av[i] = 0.f; }

            for (int kb = 0; kb < EMB; kb += 32) {
                // stage 32x16 gate & val tiles; 4 consecutive K packed per
                // ds_store_b64 (same-wave DS ops are in-order, no barrier).
#pragma unroll
                for (int j = 0; j < 4; ++j) {
                    const int k0 = half * 16 + 4 * j;
                    v4bf pg, pv;
#pragma unroll
                    for (int q = 0; q < 4; ++q) {
                        const float* rp = Wfce + (size_t)(kb + k0 + q) * HID2;
                        pg[q] = (__bf16)rp[cg + nl];
                        pv[q] = (__bf16)rp[cg + HID + nl];
                    }
                    *(v4bf*)&Blds[w][0][nl][k0] = pg;
                    *(v4bf*)&Blds[w][1][nl][k0] = pv;
                }
                const v16bf a  = ld_frag(&Alds[0][0], A_STR, lane, kb);
                const v16bf bg = ld_frag(&Blds[w][0][0][0], B_STR, lane, 0);
                const v16bf bv = ld_frag(&Blds[w][1][0][0], B_STR, lane, 0);
                ag = wmma_bf16(a, bg, ag);
                av = wmma_bf16(a, bv, av);
            }
#pragma unroll
            for (int i2 = 0; i2 < 8; ++i2) {       // C layout: m = i2 + half*8
                const int m = i2 + half * 8;
                const float g  = ag[i2];
                const float sv = g * FAST_RCP(1.f + __expf(-g)) * av[i2];
                ACT[m][w * 16 + nl] = (__bf16)sv;
            }
        }
        __syncthreads();

        // ---- proj partial: accD[nt] += act[:,chunk] @ Wproj[chunk, cols] ----
#pragma unroll
        for (int nt = 0; nt < 8; ++nt) {
            const int n0 = w * 128 + nt * 16;
            for (int kb2 = 0; kb2 < ccols; kb2 += 32) {
#pragma unroll
                for (int j = 0; j < 4; ++j) {
                    const int k0 = half * 16 + 4 * j;
                    v4bf p;
#pragma unroll
                    for (int q = 0; q < 4; ++q)
                        p[q] = (__bf16)Wpre[(size_t)(kc + kb2 + k0 + q) * EMB + n0 + nl];
                    *(v4bf*)&Blds[w][0][nl][k0] = p;
                }
                const v16bf a = ld_frag(&ACT[0][0], ACT_STR, lane, kb2);
                const v16bf b = ld_frag(&Blds[w][0][0][0], B_STR, lane, 0);
                accD[nt] = wmma_bf16(a, b, accD[nt]);
            }
        }
        __syncthreads();                           // before ACT is overwritten
    }

    // ---- epilogue: scale by routing weight, accumulate into y ----
#pragma unroll
    for (int nt = 0; nt < 8; ++nt) {
        const int n0 = w * 128 + nt * 16;
#pragma unroll
        for (int i2 = 0; i2 < 8; ++i2) {
            const int m = i2 + half * 8;
            const float v = accD[nt][i2] * s_w[m];
            atomicAdd(&y[(size_t)s_tok[m] * EMB + n0 + nl], v);
        }
    }
}

// ---------------------------------------------------------------------------
extern "C" void kernel_launch(void* const* d_in, const int* in_sizes, int n_in,
                              void* d_out, int out_size, void* d_ws, size_t ws_size,
                              hipStream_t stream) {
    const float* x     = (const float*)d_in[0];
    const float* Wr    = (const float*)d_in[1];
    const float* Wfc   = (const float*)d_in[2];
    const float* Wproj = (const float*)d_in[3];
    float* y = (float*)d_out;

    // workspace layout (~0.56 MB)
    const int ASSIGN = 2 * TOK_N + NEXP * 16;      // padded assignment capacity
    float* probs = (float*)d_ws;                   // TOK_N*NEXP
    float* zterm = probs + TOK_N * NEXP;           // TOK_N
    float* topw  = zterm + TOK_N;                  // 2*TOK_N
    int*   topi  = (int*)(topw + 2 * TOK_N);       // 2*TOK_N
    int*   off   = topi + 2 * TOK_N;               // NEXP+1
    int*   fill  = off + (NEXP + 1);               // NEXP
    int*   atok  = fill + NEXP;                    // ASSIGN
    float* aw    = (float*)(atok + ASSIGN);        // ASSIGN

    hipMemsetAsync(y,    0, (size_t)TOK_N * EMB * sizeof(float), stream);
    hipMemsetAsync(atok, 0, (size_t)ASSIGN * sizeof(int),   stream);
    hipMemsetAsync(aw,   0, (size_t)ASSIGN * sizeof(float), stream);

    k1_router <<<TOK_N / 8,   256, 0, stream>>>(x, Wr, probs, zterm, topi, topw);
    k2_reduce <<<1,           256, 0, stream>>>(probs, zterm, topi,
                                                y + (size_t)TOK_N * EMB, off, fill);
    k3_scatter<<<TOK_N / 256, 256, 0, stream>>>(topi, topw, off, fill, atok, aw);

    // allow >64KB dynamic LDS for the fused FFN kernel (320KB/WGP on CDNA5)
    (void)hipFuncSetAttribute((const void*)k4_ffn,
                              hipFuncAttributeMaxDynamicSharedMemorySize, SM_TOTAL);
    const int n_tiles = ASSIGN / 16;               // 1032 blocks (uniform early-out)
    k4_ffn    <<<n_tiles, 256, SM_TOTAL, stream>>>(x, Wfc, Wproj, atok, aw, off, y);
}
